// ProjectionRelBasis_28192165331007
// MI455X (gfx1250) — matrix operand order repaired and checked by reading, not compiled, not run on hardware
//
#include <hip/hip_runtime.h>

#define B_ROWS 2048
#define HID    1000
#define NBASE  100
#define KPAD   1024
#define LN_EPS 1e-5f

#define MT 128      // batch rows per workgroup
#define NT 64       // output cols per workgroup
#define KC 64       // K chunk = two bf16 wmma depths
#define A_STRIDE 72 // bf16 elems per LDS row (64 + pad) -> 144B, 16B aligned
#define W_STRIDE 72

typedef __attribute__((ext_vector_type(16))) __bf16 v16bf;
typedef __attribute__((ext_vector_type(8)))  float  v8f;

static __device__ __forceinline__ unsigned short f32_to_bf16(float f) {
    unsigned u = __builtin_bit_cast(unsigned, f);
    u += 0x7FFFu + ((u >> 16) & 1u);   // round-to-nearest-even
    return (unsigned short)(u >> 16);
}

union FragU {
    uint4 q[2];
    v16bf v;
};

// ---- Kernel 1: e (f32, [2048,1000]) -> padded bf16 [2048,1024] in workspace
__global__ void prep_e_bf16(const float* __restrict__ e, unsigned short* __restrict__ Ebf) {
    int idx = blockIdx.x * 256 + threadIdx.x;
    int b = idx >> 10;
    int i = idx & (KPAD - 1);
    float v = (i < HID) ? e[b * HID + i] : 0.f;
    Ebf[idx] = f32_to_bf16(v);
}

// ---- Kernel 2: the big contraction.
// accT[b,o] = sum_r att[b,r] * ( sum_i e[b,i] * rel_base[r,i,o] )
__global__ __launch_bounds__(256)
void relbasis_gemm(const unsigned short* __restrict__ Ebf,
                   const int* __restrict__ rid,
                   const float* __restrict__ rel_base,
                   const float* __restrict__ rel_att,
                   float* __restrict__ C) {
    __shared__ unsigned short Alds[MT * A_STRIDE];  // 18.0 KB
    __shared__ unsigned short Wlds[NT * W_STRIDE];  //  9.0 KB (transposed: [n][k])
    __shared__ float att_lds[MT];

    const int tid  = threadIdx.x;
    const int lane = tid & 31;
    const int wave = tid >> 5;
    const int wave_m = (wave & 3) * 32;   // 4 waves along M
    const int wave_n = (wave >> 2) * 32;  // 2 waves along N
    const int b0 = blockIdx.y * MT;
    const int n0 = blockIdx.x * NT;

    // staging roles
    const int a_row  = tid >> 1;         // 0..127
    const int a_half = tid & 1;          // which 64B half of the 128B row
    const int w_kp   = tid >> 3;         // 0..31 : k-pair (covers k=2kp, 2kp+1)
    const int w_nb   = (tid & 7) * 8;    // 0..56

    // fragment lane decomposition
    const int fm  = lane & 15;
    const int fhs = lane >> 4;

    // per-thread constant staging addresses
    const unsigned a_lds = (unsigned)(size_t)(Alds + a_row * A_STRIDE + a_half * 32);
    const size_t   a_gbase = (size_t)(b0 + a_row) * KPAD + a_half * 32;

    const v8f vzero = {0.f,0.f,0.f,0.f,0.f,0.f,0.f,0.f};
    v8f accT00 = vzero, accT01 = vzero, accT10 = vzero, accT11 = vzero;

    for (int r = 0; r < NBASE; ++r) {
        __syncthreads();  // previous att_lds readers done
        if (tid < MT)
            att_lds[tid] = rel_att[(size_t)rid[b0 + tid] * NBASE + r];

        v8f acc00 = vzero, acc01 = vzero, acc10 = vzero, acc11 = vzero;

        for (int kc = 0; kc < KPAD / KC; ++kc) {
            const int i0 = kc * KC;
            __syncthreads();  // previous tile readers done

            // ---- stage A tile (128 x 64 bf16): async copy global -> LDS (ASYNCcnt)
            {
                unsigned long long g =
                    (unsigned long long)(size_t)(Ebf + a_gbase + i0);
                asm volatile(
                    "global_load_async_to_lds_b128 %0, %1, off\n\t"
                    "global_load_async_to_lds_b128 %0, %1, off offset:16\n\t"
                    "global_load_async_to_lds_b128 %0, %1, off offset:32\n\t"
                    "global_load_async_to_lds_b128 %0, %1, off offset:48"
                    :: "v"(a_lds), "v"(g) : "memory");
            }
            // ---- stage W tile: rel_base[r, i0+k, n0..n0+63] f32 -> bf16,
            //      transposed [n][k], K-pairs packed so stores are b32.
            {
                const int k0 = i0 + 2 * w_kp;
                const int k0c = (k0 < HID) ? k0 : (HID - 1);
                const int k1c = (k0 + 1 < HID) ? (k0 + 1) : (HID - 1);
                const float* s0 = rel_base + ((size_t)r * HID + k0c) * (size_t)HID;
                const float* s1 = rel_base + ((size_t)r * HID + k1c) * (size_t)HID;
                float f0[8], f1[8];
                if ((k0 + 1) < HID && (n0 + NT) <= HID) {
                    const float4* p0 = reinterpret_cast<const float4*>(s0 + n0 + w_nb);
                    const float4* p1 = reinterpret_cast<const float4*>(s1 + n0 + w_nb);
                    float4 a0 = p0[0], a1 = p0[1], b0v = p1[0], b1v = p1[1];
                    f0[0]=a0.x; f0[1]=a0.y; f0[2]=a0.z; f0[3]=a0.w;
                    f0[4]=a1.x; f0[5]=a1.y; f0[6]=a1.z; f0[7]=a1.w;
                    f1[0]=b0v.x; f1[1]=b0v.y; f1[2]=b0v.z; f1[3]=b0v.w;
                    f1[4]=b1v.x; f1[5]=b1v.y; f1[6]=b1v.z; f1[7]=b1v.w;
                    if (k0 + KC < HID)
                        __builtin_prefetch(s0 + (size_t)KC * HID + n0 + w_nb, 0, 1);
                } else {
                    // branch-free tail: clamped addresses + select-zero
#pragma unroll
                    for (int j = 0; j < 8; ++j) {
                        int nj = n0 + w_nb + j;
                        int njc = (nj < HID) ? nj : (HID - 1);
                        float v0 = s0[njc];
                        float v1 = s1[njc];
                        bool nv = nj < HID;
                        f0[j] = (nv && k0 < HID)     ? v0 : 0.f;
                        f1[j] = (nv && k0 + 1 < HID) ? v1 : 0.f;
                    }
                }
#pragma unroll
                for (int j = 0; j < 8; ++j) {
                    unsigned pk = (unsigned)f32_to_bf16(f0[j]) |
                                  ((unsigned)f32_to_bf16(f1[j]) << 16);
                    *reinterpret_cast<unsigned*>(Wlds + (w_nb + j) * W_STRIDE + 2 * w_kp) = pk;
                }
            }
            asm volatile("s_wait_asynccnt 0x0" ::: "memory");  // async A-tile landed
            __syncthreads();

            // ---- load fragments per ISA VGPR layouts, two K-subtiles (ks)
            FragU a[2][2];   // [m-tile][ks]
            FragU w[2][2];   // [n-tile][ks]
#pragma unroll
            for (int tm = 0; tm < 2; ++tm)
#pragma unroll
                for (int ks = 0; ks < 2; ++ks) {
                    const unsigned short* ar =
                        Alds + (wave_m + tm * 16 + fm) * A_STRIDE + ks * 32 + fhs * 8;
                    a[tm][ks].q[0] = *reinterpret_cast<const uint4*>(ar);
                    a[tm][ks].q[1] = *reinterpret_cast<const uint4*>(ar + 16);
                }
#pragma unroll
            for (int tn = 0; tn < 2; ++tn)
#pragma unroll
                for (int ks = 0; ks < 2; ++ks) {
                    const unsigned short* br =
                        Wlds + (wave_n + tn * 16 + fm) * W_STRIDE + ks * 32 + fhs * 8;
                    w[tn][ks].q[0] = *reinterpret_cast<const uint4*>(br);
                    w[tn][ks].q[1] = *reinterpret_cast<const uint4*>(br + 16);
                }

            acc00 = __builtin_amdgcn_wmma_f32_16x16x32_bf16(false, a[0][0].v, false, w[0][0].v,
                                                            (short)0, acc00, false, false);
            acc01 = __builtin_amdgcn_wmma_f32_16x16x32_bf16(false, a[0][0].v, false, w[1][0].v,
                                                            (short)0, acc01, false, false);
            acc10 = __builtin_amdgcn_wmma_f32_16x16x32_bf16(false, a[1][0].v, false, w[0][0].v,
                                                            (short)0, acc10, false, false);
            acc11 = __builtin_amdgcn_wmma_f32_16x16x32_bf16(false, a[1][0].v, false, w[1][0].v,
                                                            (short)0, acc11, false, false);
            acc00 = __builtin_amdgcn_wmma_f32_16x16x32_bf16(false, a[0][1].v, false, w[0][1].v,
                                                            (short)0, acc00, false, false);
            acc01 = __builtin_amdgcn_wmma_f32_16x16x32_bf16(false, a[0][1].v, false, w[1][1].v,
                                                            (short)0, acc01, false, false);
            acc10 = __builtin_amdgcn_wmma_f32_16x16x32_bf16(false, a[1][1].v, false, w[0][1].v,
                                                            (short)0, acc10, false, false);
            acc11 = __builtin_amdgcn_wmma_f32_16x16x32_bf16(false, a[1][1].v, false, w[1][1].v,
                                                            (short)0, acc11, false, false);
        }

        // ---- fold: accT += att[b, r] * accR  (C layout: VGPR v -> M = v + 8*(lane>=16))
        const int voff = fhs * 8;
#pragma unroll
        for (int v = 0; v < 8; ++v) {
            float am0 = att_lds[wave_m + v + voff];
            float am1 = att_lds[wave_m + 16 + v + voff];
            accT00[v] += am0 * acc00[v];
            accT01[v] += am0 * acc01[v];
            accT10[v] += am1 * acc10[v];
            accT11[v] += am1 * acc11[v];
        }
    }

    // ---- store to padded C [2048, 1024] f32
    {
        const int col  = lane & 15;
        const int voff = fhs * 8;
#pragma unroll
        for (int v = 0; v < 8; ++v) {
            int r0 = b0 + wave_m + v + voff;
            int r1 = r0 + 16;
            int c0 = n0 + wave_n + col;
            int c1 = c0 + 16;
            C[(size_t)r0 * KPAD + c0] = accT00[v];
            C[(size_t)r0 * KPAD + c1] = accT01[v];
            C[(size_t)r1 * KPAD + c0] = accT10[v];
            C[(size_t)r1 * KPAD + c1] = accT11[v];
        }
    }
}

// ---- Kernel 3: bias (att @ rel_bias) + LayerNorm + sigmoid, one block per row
__global__ __launch_bounds__(256)
void relbasis_epilogue(const float* __restrict__ C,
                       const int* __restrict__ rid,
                       const float* __restrict__ rel_att,
                       const float* __restrict__ rel_bias,
                       float* __restrict__ out) {
    const int b = blockIdx.x;
    const int tid = threadIdx.x;
    __shared__ float att_s[NBASE];
    __shared__ float rs[256];
    __shared__ float rss[256];

    if (tid < NBASE)
        att_s[tid] = rel_att[(size_t)rid[b] * NBASE + tid];
    __syncthreads();

    float x[4];
    float s = 0.f, ss = 0.f;
#pragma unroll
    for (int j = 0; j < 4; ++j) {
        int o = tid + j * 256;
        float xv = 0.f;
        if (o < HID) {
            float bias = 0.f;
            for (int r = 0; r < NBASE; ++r)
                bias += att_s[r] * rel_bias[r * HID + o];
            xv = C[(size_t)b * KPAD + o] + bias;
            s += xv;
            ss += xv * xv;
        }
        x[j] = xv;
    }
    rs[tid] = s;
    rss[tid] = ss;
    __syncthreads();
    for (int off = 128; off > 0; off >>= 1) {
        if (tid < off) { rs[tid] += rs[tid + off]; rss[tid] += rss[tid + off]; }
        __syncthreads();
    }
    const float inv = 1.0f / (float)HID;
    const float mean = rs[0] * inv;
    const float var  = rss[0] * inv - mean * mean;
    const float rstd = rsqrtf(var + LN_EPS);
#pragma unroll
    for (int j = 0; j < 4; ++j) {
        int o = tid + j * 256;
        if (o < HID) {
            float y = (x[j] - mean) * rstd;
            out[(size_t)b * HID + o] = 1.0f / (1.0f + __expf(-y));
        }
    }
}

extern "C" void kernel_launch(void* const* d_in, const int* in_sizes, int n_in,
                              void* d_out, int out_size, void* d_ws, size_t ws_size,
                              hipStream_t stream) {
    const float* e        = (const float*)d_in[0];
    const int*   rid      = (const int*)d_in[1];
    const float* rel_base = (const float*)d_in[2];
    const float* rel_bias = (const float*)d_in[3];
    const float* rel_att  = (const float*)d_in[4];
    float* out = (float*)d_out;

    // workspace layout: [0, 4MB) padded bf16 E; [4MB, 12MB) padded f32 C
    unsigned short* Ebf = (unsigned short*)d_ws;
    float* C = (float*)((char*)d_ws + (size_t)B_ROWS * KPAD * sizeof(unsigned short));

    prep_e_bf16<<<(B_ROWS * KPAD) / 256, 256, 0, stream>>>(e, Ebf);
    relbasis_gemm<<<dim3(KPAD / NT, B_ROWS / MT), 256, 0, stream>>>(Ebf, rid, rel_base, rel_att, C);
    relbasis_epilogue<<<B_ROWS, 256, 0, stream>>>(C, rid, rel_att, rel_bias, out);
}
